// KeypointPatchModel_35296041238939
// MI455X (gfx1250) — matrix operand order repaired and checked by reading, not compiled, not run on hardware
//
#include <hip/hip_runtime.h>
#include <hip/hip_bf16.h>

#define HH 224
#define WW 224
#define BB 32
#define CC 3
#define KPT 64
#define PP 16
#define NCLS 1000
#define IMG (HH*WW)
#define FC1_K 2048   // 8*16*16
#define FC2_K 4096   // 64*64
#define MROWS 2048   // B*K = 32*64
#define CHUNKS 256
#define CHLEN  (IMG / CHUNKS)   // 196

typedef __attribute__((ext_vector_type(16))) _Float16 v16h;
typedef __attribute__((ext_vector_type(8)))  _Float16 v8h;
typedef __attribute__((ext_vector_type(8)))  float    v8f;

// ---------------------------------------------------------------------------
// 1) Harris response, LDS-tiled: gray(20x20 halo) -> dx/dy(18x18) -> R(16x16)
// ---------------------------------------------------------------------------
__global__ void harris_kernel(const float* __restrict__ x, float* __restrict__ R) {
    __shared__ float gsh[20][20];
    __shared__ float dxs[18][18];
    __shared__ float dys[18][18];
    const int b   = blockIdx.z;
    const int ty0 = blockIdx.y * 16, tx0 = blockIdx.x * 16;
    const int tx  = threadIdx.x, ty = threadIdx.y;
    const int t   = ty * 16 + tx;
    const float* xb = x + (size_t)b * CC * IMG;

    for (int i = t; i < 400; i += 256) {
        int ly = i / 20, lx = i % 20;
        int gy = ty0 + ly - 2, gx = tx0 + lx - 2;
        float v = 0.0f;
        if (gy >= 0 && gy < HH && gx >= 0 && gx < WW) {
            int o = gy * WW + gx;
            v = (xb[o] + xb[IMG + o] + xb[2 * IMG + o]) * (1.0f / 3.0f);
        }
        gsh[ly][lx] = v;
    }
    __syncthreads();

    for (int i = t; i < 324; i += 256) {
        int ly = i / 18, lx = i % 18;
        int gy = ty0 + ly - 1, gx = tx0 + lx - 1;
        float dx = 0.0f, dy = 0.0f;
        if (gy >= 0 && gy < HH && gx >= 0 && gx < WW) {
            float a00 = gsh[ly][lx],     a01 = gsh[ly][lx + 1],     a02 = gsh[ly][lx + 2];
            float a10 = gsh[ly + 1][lx],                            a12 = gsh[ly + 1][lx + 2];
            float a20 = gsh[ly + 2][lx], a21 = gsh[ly + 2][lx + 1], a22 = gsh[ly + 2][lx + 2];
            dx = (a02 + a12 + a22) - (a00 + a10 + a20);   // cross-correlation, dxk
            dy = (a20 + a21 + a22) - (a00 + a01 + a02);   // dyk
        }
        dxs[ly][lx] = dx;
        dys[ly][lx] = dy;
    }
    __syncthreads();

    const int gy = ty0 + ty, gx = tx0 + tx;
    const float gw1[3] = {0.25f, 0.5f, 0.25f};   // outer product -> gauss/16
    float Sxx = 0.f, Syy = 0.f, Sxy = 0.f;
#pragma unroll
    for (int r = 0; r < 3; ++r) {
#pragma unroll
        for (int c = 0; c < 3; ++c) {
            float fw = gw1[r] * gw1[c];
            float dv = dxs[ty + r][tx + c];
            float ev = dys[ty + r][tx + c];
            Sxx += fw * dv * dv;
            Syy += fw * ev * ev;
            Sxy += fw * dv * ev;
        }
    }
    float tr = Sxx + Syy;
    float Rv = Sxx * Syy - Sxy * Sxy - 0.04f * tr * tr;
    bool inmask = (gy >= PP && gy < HH - PP && gx >= PP && gx < WW - PP);
    R[(size_t)b * IMG + gy * WW + gx] = inmask ? Rv : 0.0f;
}

// ---------------------------------------------------------------------------
// 2) Top-64 per image: chunked incremental argmax.
//    Chunk c = { i : i % 256 == c } (phase-1 fully coalesced). Per round:
//    reduce 256 LDS chunk-maxima, then only the winning chunk's owner thread
//    rescans its 196 elements. (v,globalIdx) pairs keep JAX top_k ordering
//    (descending, lowest-index ties). R is rewritten each call -> determinism.
// ---------------------------------------------------------------------------
__global__ void topk_kernel(float* __restrict__ R, float* __restrict__ coords) {
    __shared__ float cmax[CHUNKS];
    __shared__ int   cidx[CHUNKS];
    __shared__ float rv[CHUNKS];
    __shared__ int   ri[CHUNKS];
    const int b = blockIdx.x;
    const int t = threadIdx.x;
    float* Rb = R + (size_t)b * IMG;

    float best = -__builtin_inff();
    int   bi   = 0x7FFFFFFF;
    for (int j = 0; j < CHLEN; ++j) {
        int i = t + CHUNKS * j;          // coalesced across the block
        float v = Rb[i];
        if (v > best || (v == best && i < bi)) { best = v; bi = i; }
    }
    cmax[t] = best; cidx[t] = bi;
    __syncthreads();

    for (int kp = 0; kp < KPT; ++kp) {
        rv[t] = cmax[t]; ri[t] = cidx[t];
        __syncthreads();
        for (int s = 128; s > 0; s >>= 1) {
            if (t < s) {
                float v2 = rv[t + s]; int i2 = ri[t + s];
                if (v2 > rv[t] || (v2 == rv[t] && i2 < ri[t])) { rv[t] = v2; ri[t] = i2; }
            }
            __syncthreads();
        }
        const int gidx = ri[0];
        if (t == 0) {
            int row = gidx / WW, col = gidx % WW;
            float yn = ((float)row / (float)HH - 0.5f) * 2.0f;
            float xn = ((float)col / (float)WW - 0.5f) * 2.0f;
            yn = fminf(fmaxf(yn, -1.0f), 1.0f);
            xn = fminf(fmaxf(xn, -1.0f), 1.0f);
            coords[((size_t)b * KPT + kp) * 2 + 0] = yn;
            coords[((size_t)b * KPT + kp) * 2 + 1] = xn;
        }
        // winning chunk's owner rescans its chunk (all its stores/loads are
        // same-thread -> ISA-ordered; guard current gidx explicitly too)
        if (t == (gidx & (CHUNKS - 1))) {
            Rb[gidx] = -__builtin_inff();
            float nb = -__builtin_inff(); int nbi = 0x7FFFFFFF;
            for (int j = 0; j < CHLEN; ++j) {
                int i = t + CHUNKS * j;
                float v = (i == gidx) ? -__builtin_inff() : Rb[i];
                if (v > nb || (v == nb && i < nbi)) { nb = v; nbi = i; }
            }
            cmax[t] = nb; cidx[t] = nbi;
        }
        __syncthreads();
    }
}

// ---------------------------------------------------------------------------
// 3) Bilinear patch extraction (faithful to reference's x/y-swapped broadcast)
// ---------------------------------------------------------------------------
__device__ __forceinline__ float samp_img(const float* __restrict__ xb, float xf, float yf) {
    if (xf < 0.0f || xf >= (float)WW || yf < 0.0f || yf >= (float)HH) return 0.0f;
    int xi = (int)xf, yi = (int)yf;
    return xb[yi * WW + xi];
}

__global__ void patch_kernel(const float* __restrict__ x, const float* __restrict__ coords,
                             float* __restrict__ patches) {
    const int bk = blockIdx.x;           // 0..2047 = b*64 + kp
    const int c  = blockIdx.y;           // channel
    const int t  = threadIdx.x;          // py*16+px
    const int py = t / PP, px = t % PP;
    const int b  = bk >> 6;

    float cy01 = (coords[(size_t)bk * 2 + 0] + 1.0f) * 0.5f;  // row component
    float cx01 = (coords[(size_t)bk * 2 + 1] + 1.0f) * 0.5f;  // col component
    const float r = (float)PP / (float)HH;
    float pgx = -r + (2.0f * r) * (float)px / 15.0f;
    float pgy = -r + (2.0f * r) * (float)py / 15.0f;
    // reference adds (x_off,y_off) to (y01,x01): gx uses cy01, gy uses cx01
    float gx = (pgx + cy01) * (float)(WW - 1);
    float gy = (pgy + cx01) * (float)(HH - 1);

    float x0 = floorf(gx), y0 = floorf(gy);
    float wx1 = gx - x0, wx0 = 1.0f - wx1;
    float wy1 = gy - y0, wy0 = 1.0f - wy1;

    const float* xb = x + ((size_t)b * CC + c) * IMG;
    float v = samp_img(xb, x0,        y0)        * wx0 * wy0
            + samp_img(xb, x0 + 1.0f, y0)        * wx1 * wy0
            + samp_img(xb, x0,        y0 + 1.0f) * wx0 * wy1
            + samp_img(xb, x0 + 1.0f, y0 + 1.0f) * wx1 * wy1;
    patches[((size_t)bk * CC + c) * (PP * PP) + t] = v;
}

// ---------------------------------------------------------------------------
// 4) f32 -> f16 cast (weights)
// ---------------------------------------------------------------------------
__global__ void cast_f16_kernel(const float* __restrict__ src, _Float16* __restrict__ dst, int n) {
    int i = blockIdx.x * blockDim.x + threadIdx.x;
    if (i < n) dst[i] = (_Float16)src[i];
}

// ---------------------------------------------------------------------------
// 5) Per-patch conv 3->8 (3x3, SAME) + bias + ReLU -> f16 GEMM A matrix.
//    One block per patch; input window cached in registers, all 8 co's looped.
// ---------------------------------------------------------------------------
__global__ void conv_kernel(const float* __restrict__ patches, const float* __restrict__ conv_w,
                            const float* __restrict__ conv_b, _Float16* __restrict__ actA) {
    __shared__ float sp[CC][PP * PP];
    __shared__ float sw[8][CC][3][3];   // 216 floats
    __shared__ float sb[8];
    const int bk = blockIdx.x;
    const int t  = threadIdx.x;

    for (int i = t; i < CC * PP * PP; i += 256)
        sp[i / (PP * PP)][i % (PP * PP)] = patches[(size_t)bk * (CC * PP * PP) + i];
    if (t < 216) ((float*)sw)[t] = conv_w[t];
    if (t < 8)   sb[t] = conv_b[t];
    __syncthreads();

    const int py = t / PP, px = t % PP;
    float win[CC][3][3];
#pragma unroll
    for (int ci = 0; ci < CC; ++ci)
#pragma unroll
        for (int r = 0; r < 3; ++r)
#pragma unroll
            for (int c = 0; c < 3; ++c) {
                int yy = py + r - 1, xx = px + c - 1;
                win[ci][r][c] = (yy >= 0 && yy < PP && xx >= 0 && xx < PP)
                                    ? sp[ci][yy * PP + xx] : 0.0f;
            }

#pragma unroll
    for (int co = 0; co < 8; ++co) {
        float acc = sb[co];
#pragma unroll
        for (int ci = 0; ci < CC; ++ci)
#pragma unroll
            for (int r = 0; r < 3; ++r)
#pragma unroll
                for (int c = 0; c < 3; ++c)
                    acc += sw[co][ci][r][c] * win[ci][r][c];
        acc = acc > 0.0f ? acc : 0.0f;
        actA[(size_t)bk * FC1_K + co * (PP * PP) + t] = (_Float16)acc;
    }
}

// ---------------------------------------------------------------------------
// WMMA fragment helpers (wave32; one wave per 16x16 tile).
// A: lanes 0-15 M=0..15 K-base 0, lanes 16-31 K-base 16; elem e -> K=base+e
//    (two contiguous 16B loads).
// B stored as W[n][k] row-major: lane n=L%16, K = 16*(L/16)+e (32B contiguous).
// ---------------------------------------------------------------------------
__device__ __forceinline__ v16h load_a_frag(const _Float16* __restrict__ rowbase, int lda) {
    int lane = threadIdx.x & 31;
    const _Float16* p = rowbase + (size_t)(lane & 15) * lda + 16 * (lane >> 4);
    v8h lo = *(const v8h*)(p);
    v8h hi = *(const v8h*)(p + 8);
    v16h a;
#pragma unroll
    for (int e = 0; e < 8; ++e) { a[e] = lo[e]; a[8 + e] = hi[e]; }
    return a;
}

__device__ __forceinline__ v16h load_b_frag(const _Float16* __restrict__ rowbase, int ldb) {
    int lane = threadIdx.x & 31;
    const _Float16* p = rowbase + (size_t)(lane & 15) * ldb + 16 * (lane >> 4);
    v8h lo = *(const v8h*)(p);
    v8h hi = *(const v8h*)(p + 8);
    v16h b;
#pragma unroll
    for (int e = 0; e < 8; ++e) { b[e] = lo[e]; b[8 + e] = hi[e]; }
    return b;
}

// ---------------------------------------------------------------------------
// 6) FC1 GEMM: [2048 x 2048] x [2048 x 64] via v_wmma_f32_16x16x32_f16,
//    + bias + ReLU, scatter into h2 [32 x 4096] f16 (col = kp*64 + feat)
// ---------------------------------------------------------------------------
__global__ void fc1_wmma_kernel(const _Float16* __restrict__ A, const _Float16* __restrict__ W1,
                                const float* __restrict__ fc1_b, _Float16* __restrict__ h2) {
    const int tileM = blockIdx.x;   // 0..127
    const int tileN = blockIdx.y;   // 0..3
    const _Float16* Abase = A  + (size_t)(tileM * 16) * FC1_K;
    const _Float16* Bbase = W1 + (size_t)(tileN * 16) * FC1_K;
    v8f acc = {};
    for (int k0 = 0; k0 < FC1_K; k0 += 32) {
        if (k0 + 32 < FC1_K) {
            __builtin_prefetch(Abase + ((threadIdx.x & 15) * FC1_K) + k0 + 32, 0, 3);
            __builtin_prefetch(Bbase + ((threadIdx.x & 15) * FC1_K) + k0 + 32, 0, 3);
        }
        v16h a = load_a_frag(Abase + k0, FC1_K);
        v16h b = load_b_frag(Bbase + k0, FC1_K);
        acc = __builtin_amdgcn_wmma_f32_16x16x32_f16(false, a, false, b, (short)0, acc, false, false);
    }
    const int lane = threadIdx.x & 31;
    const int g = lane >> 4, nl = lane & 15;
    const int n = tileN * 16 + nl;
    const float bias = fc1_b[n];
#pragma unroll
    for (int r = 0; r < 8; ++r) {
        int m = tileM * 16 + 8 * g + r;
        float v = acc[r] + bias;
        v = v > 0.0f ? v : 0.0f;
        int bimg = m >> 6;     // /64
        int kp   = m & 63;
        h2[(size_t)bimg * FC2_K + kp * 64 + n] = (_Float16)v;
    }
}

// ---------------------------------------------------------------------------
// 7) FC2 GEMM: [32 x 4096] x [4096 x 1000] via WMMA, + bias -> out f32
// ---------------------------------------------------------------------------
__global__ void fc2_wmma_kernel(const _Float16* __restrict__ h2, const _Float16* __restrict__ W2,
                                const float* __restrict__ fc2_b, float* __restrict__ out) {
    const int tileN = blockIdx.x;   // 0..62  (63*16 = 1008 >= 1000)
    const int tileM = blockIdx.y;   // 0..1
    const int lane  = threadIdx.x & 31;
    const int g = lane >> 4, nl = lane & 15;
    const int n = tileN * 16 + nl;
    const _Float16* Abase = h2 + (size_t)(tileM * 16) * FC2_K;
    const _Float16* Brow  = W2 + (size_t)n * FC2_K;   // row n of W2 (contiguous in K)
    v8f acc = {};
    for (int k0 = 0; k0 < FC2_K; k0 += 32) {
        if (k0 + 32 < FC2_K) {
            __builtin_prefetch(Abase + ((lane & 15) * FC2_K) + k0 + 32, 0, 3);
        }
        v16h a = load_a_frag(Abase + k0, FC2_K);
        v16h b;
        if (n < NCLS) {
            const _Float16* p = Brow + k0 + 16 * g;
            v8h lo = *(const v8h*)(p);
            v8h hi = *(const v8h*)(p + 8);
#pragma unroll
            for (int e = 0; e < 8; ++e) { b[e] = lo[e]; b[8 + e] = hi[e]; }
        } else {
#pragma unroll
            for (int e = 0; e < 16; ++e) b[e] = (_Float16)0.0f;
        }
        // branch re-converged: EXEC all-ones at the WMMA
        acc = __builtin_amdgcn_wmma_f32_16x16x32_f16(false, a, false, b, (short)0, acc, false, false);
    }
    if (n < NCLS) {
        float bias = fc2_b[n];
#pragma unroll
        for (int r = 0; r < 8; ++r) {
            int m = tileM * 16 + 8 * g + r;   // m < 32
            out[(size_t)m * NCLS + n] = acc[r] + bias;
        }
    }
}

// ---------------------------------------------------------------------------
// Launch
// ---------------------------------------------------------------------------
extern "C" void kernel_launch(void* const* d_in, const int* in_sizes, int n_in,
                              void* d_out, int out_size, void* d_ws, size_t ws_size,
                              hipStream_t stream) {
    (void)in_sizes; (void)n_in; (void)out_size; (void)ws_size;
    const float* x      = (const float*)d_in[0];
    const float* conv_w = (const float*)d_in[1];
    const float* conv_b = (const float*)d_in[2];
    const float* fc1_w  = (const float*)d_in[3];
    const float* fc1_b  = (const float*)d_in[4];
    const float* fc2_w  = (const float*)d_in[5];
    const float* fc2_b  = (const float*)d_in[6];
    float* out = (float*)d_out;

    char* ws = (char*)d_ws;
    size_t off = 0;
    auto wsalloc = [&](size_t bytes) -> void* {
        void* p = ws + off;
        off = (off + bytes + 255) & ~(size_t)255;
        return p;
    };
    float*    R       = (float*)   wsalloc(sizeof(float) * BB * IMG);
    float*    coords  = (float*)   wsalloc(sizeof(float) * BB * KPT * 2);
    float*    patches = (float*)   wsalloc(sizeof(float) * (size_t)MROWS * CC * PP * PP);
    _Float16* actA    = (_Float16*)wsalloc(sizeof(_Float16) * (size_t)MROWS * FC1_K);
    _Float16* w1h     = (_Float16*)wsalloc(sizeof(_Float16) * 64 * FC1_K);
    _Float16* w2h     = (_Float16*)wsalloc(sizeof(_Float16) * (size_t)NCLS * FC2_K);
    _Float16* h2      = (_Float16*)wsalloc(sizeof(_Float16) * BB * FC2_K);

    harris_kernel<<<dim3(WW / 16, HH / 16, BB), dim3(16, 16, 1), 0, stream>>>(x, R);
    topk_kernel<<<BB, CHUNKS, 0, stream>>>(R, coords);
    patch_kernel<<<dim3(MROWS, CC), 256, 0, stream>>>(x, coords, patches);
    {
        int n1 = 64 * FC1_K;
        cast_f16_kernel<<<(n1 + 255) / 256, 256, 0, stream>>>(fc1_w, w1h, n1);
        int n2 = NCLS * FC2_K;
        cast_f16_kernel<<<(n2 + 255) / 256, 256, 0, stream>>>(fc2_w, w2h, n2);
    }
    conv_kernel<<<MROWS, 256, 0, stream>>>(patches, conv_w, conv_b, actA);
    fc1_wmma_kernel<<<dim3(MROWS / 16, 64 / 16), 32, 0, stream>>>(actA, w1h, fc1_b, h2);
    fc2_wmma_kernel<<<dim3(63, 2), 32, 0, stream>>>(h2, w2h, fc2_b, out);
}
